// EdgeGraphAttentionLayer_730144441121
// MI455X (gfx1250) — matrix operand order repaired and checked by reading, compiled, not run on hardware
//
#include <hip/hip_runtime.h>
#include <hip/hip_bf16.h>

// ---------------------------------------------------------------------------
// EdgeGraphAttentionLayer, algebraically factored for MI455X (gfx1250).
//   N=512, DIN=256, EIN=64, DOUT=128, EOUT=64
// Key identity: new_hij[i,j] = h_src[i] + h_e[i,j] + h_dst[j]  =>
//   scores[i,j]   = s_src[i] + s_dst[j] + e[ij] . (We@att)
//   new_edge[i,j] = e[ij] @ (We@ET) + (h_src@ET)[i] + (h_dst@ET)[j]
//   new_x[i]      = h_src[i] + (attn[i] @ E_i) @ We + (attn @ h_dst)[i]
// so the 128MB new_hij tensor is never materialized. Bandwidth-bound:
// ~2 reads of e (128MB) + 64MB edge write ~= 9us at 23.3 TB/s.
// K5 stages E_i through LDS with the Tensor Data Mover (tensor_load_to_lds),
// then feeds v_wmma_f32_16x16x32_bf16 and the attention aggregation from the
// same on-chip copy.
// ---------------------------------------------------------------------------

typedef __attribute__((ext_vector_type(16))) __bf16 v16bf;
typedef __attribute__((ext_vector_type(8)))  float  v8f;
typedef __attribute__((ext_vector_type(2)))  float  v2f;
typedef __attribute__((ext_vector_type(4)))  unsigned int v4u;
typedef __attribute__((ext_vector_type(8)))  int    v8i;
typedef __attribute__((ext_vector_type(4)))  int    v4i;

#define NN    512
#define DIN   256
#define EIN   64
#define DOUT  128
#define EOUT  64
#define LALPHA 0.2f

static __device__ __forceinline__ unsigned short f32_to_bf16_bits(float f) {
  union { float f; unsigned u; } v; v.f = f;
  unsigned r = v.u + 0x7FFFu + ((v.u >> 16) & 1u);   // round-to-nearest-even
  return (unsigned short)(r >> 16);
}
static __device__ __forceinline__ __bf16 bits_to_bf16(unsigned short s) {
  union { unsigned short s; __bf16 b; } o; o.s = s; return o.b;
}
static __device__ __forceinline__ __bf16 f2bf(float f) {
  return bits_to_bf16(f32_to_bf16_bits(f));
}

// ---------------------------------------------------------------------------
// TDM: async-load a contiguous run of `nelem` f32 (1-D tile) into LDS.
// D# group0: count=1 | lds_addr | global_addr[56:0] | type=2 ("image").
// D# group1: data_size=4B; tensor_dim0 = tile_dim0 = nelem; 1-D (tile_dim1=0).
// 6-arg builtin form (clang-23 / therock-10.0 headers): the two int32x4 and
// the int32x8 trailing groups are the (unused here, <=2D) D# groups 2/3.
// Tracked by TENSORcnt.
// ---------------------------------------------------------------------------
static __device__ __forceinline__ void tdm_load_f32(const float* src, void* lds,
                                                    unsigned nelem) {
  unsigned long long ga = (unsigned long long)(uintptr_t)src;
  unsigned la = (unsigned)(uintptr_t)lds;   // low 32 bits of flat LDS ptr = LDS addr
  v4u g0;
  g0[0] = 1u;                                        // count=1, user descriptor
  g0[1] = la;                                        // lds_addr
  g0[2] = (unsigned)(ga & 0xFFFFFFFFu);              // global_addr[31:0]
  g0[3] = (unsigned)((ga >> 32) & 0x1FFFFFFu) | (2u << 30);  // [56:32] | type=2
  v8i g1;
  g1[0] = (int)(2u << 16);           // wg_mask=0, data_size=2 (4B), no flags
  g1[1] = (int)(nelem << 16);        // tensor_dim0[15:0] (abar_addr=0)
  g1[2] = 0;                         // tensor_dim0[31:16]=0, tensor_dim1=0
  g1[3] = (int)(nelem << 16);        // tile_dim0 = nelem
  g1[4] = 0;                         // tile_dim1=0 (1-D), tile_dim2=0
  g1[5] = (int)nelem;                // tensor_dim0_stride
  g1[6] = 0;
  g1[7] = 0;
  v4i z4 = {0, 0, 0, 0};             // groups 2/3 unused (<=2D tensor)
  v8i z8 = {0, 0, 0, 0, 0, 0, 0, 0};
  __builtin_amdgcn_tensor_load_to_lds(g0, g1, z4, z4, z8, 0);
}

// ---------------------------------------------------------------------------
// K1: tiny precompute.  wv = We @ att (64), Mbf = bf16(We @ ET) (64x64).
// ---------------------------------------------------------------------------
__global__ void k1_small(const float* __restrict__ tt, const float* __restrict__ ET,
                         const float* __restrict__ att,
                         float* __restrict__ wv, unsigned short* __restrict__ Mbf) {
  const int t = threadIdx.x;                 // 1 block x 256 threads
  const float* We = tt + DIN * DOUT;         // rows [256, 320) of triplet_transform
  if (t < EIN) {
    float s = 0.f;
    for (int c = 0; c < DOUT; ++c) s += We[t * DOUT + c] * att[c];
    wv[t] = s;
  }
  for (int idx = t; idx < EIN * EOUT; idx += 256) {
    int k = idx / EOUT, c2 = idx % EOUT;
    float s = 0.f;
    for (int c = 0; c < DOUT; ++c) s += We[k * DOUT + c] * ET[c * EOUT + c2];
    Mbf[idx] = f32_to_bf16_bits(s);
  }
}

// ---------------------------------------------------------------------------
// K2: h_src = x @ Wa, h_dst = x @ Wb via V_WMMA_F32_16X16X4_F32 (full f32).
// One 16x16 C-tile per wave; 512 tiles total (2 matrices x 32x8 tiles).
// ---------------------------------------------------------------------------
__global__ void __launch_bounds__(128) k2_h(const float* __restrict__ x,
                                            const float* __restrict__ tt,
                                            float* __restrict__ h_src,
                                            float* __restrict__ h_dst) {
  const int lane = threadIdx.x & 31;
  const int wave = threadIdx.x >> 5;          // 0..3
  const int tile = blockIdx.x * 4 + wave;     // 0..511
  const int mat  = tile >> 8;                 // 0: Wa/h_src, 1: Wb/h_dst
  const int tr   = tile & 255;
  const int m0   = (tr >> 3) * 16;
  const int n0   = (tr & 7) * 16;
  const float* B = tt + (mat ? (DIN + EIN) * DOUT : 0);
  float* H = mat ? h_dst : h_src;
  const int h = lane >> 4, l = lane & 15;

  v8f acc = {};
  for (int k0 = 0; k0 < DIN; k0 += 4) {
    v2f a, b;
    a[0] = x[(m0 + l) * DIN + k0 + 2 * h];
    a[1] = x[(m0 + l) * DIN + k0 + 2 * h + 1];
    b[0] = B[(k0 + 2 * h) * DOUT + n0 + l];
    b[1] = B[(k0 + 2 * h + 1) * DOUT + n0 + l];
    acc = __builtin_amdgcn_wmma_f32_16x16x4_f32(false, a, false, b,
                                                (short)0, acc, false, false);
  }
#pragma unroll
  for (int r = 0; r < 8; ++r)
    H[(m0 + r + 8 * h) * DOUT + n0 + l] = acc[r];   // C layout: M=r / 8+r
}

// ---------------------------------------------------------------------------
// K3: c_{src,dst} = h_{src,dst} @ ET (512x64 each); s_{src,dst} = h @ att.
// ---------------------------------------------------------------------------
__global__ void k3_cs(const float* __restrict__ h_src, const float* __restrict__ h_dst,
                      const float* __restrict__ ET, const float* __restrict__ att,
                      float* __restrict__ c_src, float* __restrict__ c_dst,
                      float* __restrict__ s_src, float* __restrict__ s_dst) {
  const int gid = blockIdx.x * blockDim.x + threadIdx.x;  // 65536 threads
  const int sel = gid >> 15;
  const int rem = gid & 32767;
  const int i = rem >> 6, c = rem & 63;
  const float* H = sel ? h_dst : h_src;
  float* C = sel ? c_dst : c_src;
  float s = 0.f;
  for (int k = 0; k < DOUT; ++k) s += H[i * DOUT + k] * ET[k * EOUT + c];
  C[i * EOUT + c] = s;
  if (c == 0) {
    float sa = 0.f;
    for (int k = 0; k < DOUT; ++k) sa += H[i * DOUT + k] * att[k];
    (sel ? s_dst : s_src)[i] = sa;
  }
}

// ---------------------------------------------------------------------------
// K4: scores + leaky-relu + row softmax. One block per source row i.
// scores[i,j] = s_src[i] + s_dst[j] + e[ij].wv   (streams e once, 64MB,
// global_load_b128 per lane).
// ---------------------------------------------------------------------------
__global__ void __launch_bounds__(NN) k4_softmax(const float* __restrict__ e,
                                                 const float* __restrict__ wv_g,
                                                 const float* __restrict__ s_src,
                                                 const float* __restrict__ s_dst,
                                                 float* __restrict__ attn) {
  __shared__ float wv[EIN];
  __shared__ float red[NN];
  const int i = blockIdx.x, j = threadIdx.x;      // 512 threads = 16 waves
  if (j < EIN) wv[j] = wv_g[j];
  __syncthreads();

  const float4* er4 = (const float4*)(e + ((size_t)i * NN + j) * EIN);
  __builtin_prefetch(er4, 0, 0);                  // global_prefetch_b8
  float s = s_src[i] + s_dst[j];
#pragma unroll
  for (int k = 0; k < EIN / 4; ++k) {
    float4 v = er4[k];
    s += v.x * wv[4*k] + v.y * wv[4*k+1] + v.z * wv[4*k+2] + v.w * wv[4*k+3];
  }
  s = (s >= 0.f) ? s : LALPHA * s;

  red[j] = s; __syncthreads();
  for (int st = NN / 2; st > 0; st >>= 1) {
    if (j < st) red[j] = fmaxf(red[j], red[j + st]);
    __syncthreads();
  }
  const float mx = red[0]; __syncthreads();
  const float ex = __expf(s - mx);
  red[j] = ex; __syncthreads();
  for (int st = NN / 2; st > 0; st >>= 1) {
    if (j < st) red[j] += red[j + st];
    __syncthreads();
  }
  attn[i * NN + j] = ex * (1.f / red[0]);
}

// ---------------------------------------------------------------------------
// K5: fused edge-output GEMM + attention aggregation, one block per row i.
// E_i (512x64 f32) TDM-loaded into LDS in four 32KB chunks (128 rows each);
// each chunk feeds v_wmma_f32_16x16x32_bf16 (E_i @ M) AND
// agg_e[i] = attn[i] @ E_i, so e is read from HBM exactly once here.
// B-fragments of M (constant) live in registers for the whole block.
// ---------------------------------------------------------------------------
__global__ void __launch_bounds__(256) k5_edges(const float* __restrict__ e,
                                                const float* __restrict__ attn,
                                                const unsigned short* __restrict__ Mbf,
                                                const float* __restrict__ c_src,
                                                const float* __restrict__ c_dst,
                                                float* __restrict__ out_edges,
                                                float* __restrict__ agg_e) {
  __shared__ float4 eLdsV[128 * (EIN / 4)];    // 32 KB chunk of E_i (f32)
  __shared__ float  red[256];
  const float* eL = (const float*)eLdsV;

  const int i = blockIdx.x;
  const int t = threadIdx.x;
  const int lane = t & 31, wave = t >> 5;      // 8 waves
  const int h = lane >> 4, l = lane & 15;

  // Preload all eight 32x16 B-fragments of M = We@ET (block-constant).
  // B layout: Vv holds K = 16h + 2v, 16h + 2v + 1; N = l.
  v16bf Bf[8];
#pragma unroll
  for (int nt = 0; nt < 4; ++nt) {
#pragma unroll
    for (int kk = 0; kk < 2; ++kk) {
#pragma unroll
      for (int v = 0; v < 8; ++v) {
        const int kB = kk * 32 + 16 * h + 2 * v;
        Bf[nt * 2 + kk][2 * v]     = bits_to_bf16(Mbf[(kB    ) * EOUT + nt * 16 + l]);
        Bf[nt * 2 + kk][2 * v + 1] = bits_to_bf16(Mbf[(kB + 1) * EOUT + nt * 16 + l]);
      }
    }
  }

  float aggAcc = 0.f;
  const int cA = t & 63, gA = t >> 6;          // agg: column, j-group (4x32 rows)

  for (int chunk = 0; chunk < 4; ++chunk) {
    const int jBase = chunk * 128;
    __syncthreads();                           // prev chunk fully consumed
    if (t < 32) {                              // wave 0 drives the TDM
      tdm_load_f32(e + ((size_t)i * NN + jBase) * EIN, (void*)eLdsV, 128 * EIN);
      __builtin_amdgcn_s_wait_tensorcnt(0);    // s_wait_tensorcnt 0
    }
    __syncthreads();

    // Each wave owns one 16-row strip of the 128-row chunk.
    const int row0 = wave * 16;
    // A 16x32 bf16 frags for K=0..31 and K=32..63: per lane one contiguous
    // 16-f32 run (K = 16h..16h+15) -> four ds_load_b128, converted to bf16 in registers.
    v16bf Af[2];
#pragma unroll
    for (int kk = 0; kk < 2; ++kk) {
      const int fb = (row0 + l) * (EIN / 4) + kk * 8 + 4 * h;
      float4 r0 = eLdsV[fb], r1 = eLdsV[fb + 1];
      float4 r2 = eLdsV[fb + 2], r3 = eLdsV[fb + 3];
      Af[kk][0] = f2bf(r0.x); Af[kk][1] = f2bf(r0.y);
      Af[kk][2] = f2bf(r0.z); Af[kk][3] = f2bf(r0.w);
      Af[kk][4] = f2bf(r1.x); Af[kk][5] = f2bf(r1.y);
      Af[kk][6] = f2bf(r1.z); Af[kk][7] = f2bf(r1.w);
      Af[kk][8]  = f2bf(r2.x); Af[kk][9]  = f2bf(r2.y);
      Af[kk][10] = f2bf(r2.z); Af[kk][11] = f2bf(r2.w);
      Af[kk][12] = f2bf(r3.x); Af[kk][13] = f2bf(r3.y);
      Af[kk][14] = f2bf(r3.z); Af[kk][15] = f2bf(r3.w);
    }
#pragma unroll
    for (int nt = 0; nt < 4; ++nt) {
      v8f acc = {};
      acc = __builtin_amdgcn_wmma_f32_16x16x32_bf16(false, Af[0], false, Bf[nt * 2],
                                                    (short)0, acc, false, false);
      acc = __builtin_amdgcn_wmma_f32_16x16x32_bf16(false, Af[1], false, Bf[nt * 2 + 1],
                                                    (short)0, acc, false, false);
#pragma unroll
      for (int r = 0; r < 8; ++r) {
        const int j = jBase + row0 + r + 8 * h;
        const int c = nt * 16 + l;
        out_edges[((size_t)i * NN + j) * EOUT + c] =
            acc[r] + c_src[i * EOUT + c] + c_dst[j * EOUT + c];
      }
    }

    // Attention aggregation over this chunk (reuses LDS copy of E_i, f32).
    for (int jl = gA * 32; jl < gA * 32 + 32; ++jl)
      aggAcc += attn[i * NN + jBase + jl] * eL[jl * EIN + cA];
  }

  red[t] = aggAcc;
  __syncthreads();
  if (t < 64)
    agg_e[i * EOUT + t] = red[t] + red[64 + t] + red[128 + t] + red[192 + t];
}

// ---------------------------------------------------------------------------
// K6: new_x[i,c] = h_src[i,c] + agg_e[i] @ We[:,c] + attn[i] @ h_dst[:,c]
// ---------------------------------------------------------------------------
__global__ void __launch_bounds__(DOUT) k6_newx(const float* __restrict__ h_src,
                                                const float* __restrict__ h_dst,
                                                const float* __restrict__ attn,
                                                const float* __restrict__ agg_e,
                                                const float* __restrict__ tt,
                                                float* __restrict__ new_x) {
  const int i = blockIdx.x, c = threadIdx.x;   // 512 x 128
  const float* We = tt + DIN * DOUT;
  float s = h_src[i * DOUT + c];
  for (int k = 0; k < EIN; ++k) s += agg_e[i * EIN + k] * We[k * DOUT + c];
  float a = 0.f;
  for (int j = 0; j < NN; ++j) a += attn[i * NN + j] * h_dst[j * DOUT + c];
  new_x[i * DOUT + c] = s + a;
}

// ---------------------------------------------------------------------------
// Workspace layout (floats): ~1.9 MB total.
// ---------------------------------------------------------------------------
extern "C" void kernel_launch(void* const* d_in, const int* in_sizes, int n_in,
                              void* d_out, int out_size, void* d_ws, size_t ws_size,
                              hipStream_t stream) {
  (void)in_sizes; (void)n_in; (void)out_size; (void)ws_size;
  const float* x   = (const float*)d_in[0];   // (1,512,256)
  const float* e   = (const float*)d_in[1];   // (1,512*512,64)
  const float* tt  = (const float*)d_in[2];   // (576,128)
  const float* ET  = (const float*)d_in[3];   // (128,64)
  const float* att = (const float*)d_in[4];   // (128,1)

  float* out = (float*)d_out;
  float* ws  = (float*)d_ws;
  float* h_src = ws + 0;         // 512*128
  float* h_dst = ws + 65536;     // 512*128
  float* c_src = ws + 131072;    // 512*64
  float* c_dst = ws + 163840;    // 512*64
  float* s_src = ws + 196608;    // 512
  float* s_dst = ws + 197120;    // 512
  float* wv    = ws + 197632;    // 64
  float* attn  = ws + 197696;    // 512*512
  float* agg_e = ws + 459840;    // 512*64
  unsigned short* Mbf = (unsigned short*)(ws + 492608);  // 64*64 bf16

  float* new_x     = out;                        // (1,512,128) first
  float* out_edges = out + (size_t)NN * DOUT;    // (1,512*512,64) second

  k1_small  <<<1,   256, 0, stream>>>(tt, ET, att, wv, Mbf);
  k2_h      <<<128, 128, 0, stream>>>(x, tt, h_src, h_dst);
  k3_cs     <<<256, 256, 0, stream>>>(h_src, h_dst, ET, att, c_src, c_dst, s_src, s_dst);
  k4_softmax<<<NN,  NN,  0, stream>>>(e, wv, s_src, s_dst, attn);
  k5_edges  <<<NN,  256, 0, stream>>>(e, attn, Mbf, c_src, c_dst, out_edges, agg_e);
  k6_newx   <<<NN,  DOUT,0, stream>>>(h_src, h_dst, attn, agg_e, tt, new_x);
}
